// DCMMSRAttention_77610059038813
// MI455X (gfx1250) — matrix operand order repaired and checked by reading, not compiled, not run on hardware
//
#include <hip/hip_runtime.h>
#include <hip/hip_bf16.h>

// ---------------- problem constants (match reference setup_inputs) ----------
#define BATCH 2
#define SEQ   512
#define EMB   512
#define HD    8
#define DH    64          // head dim
#define WSZ   64          // window size
#define NW    8           // number of windows = SEQ/WSZ
#define NTOP  4           // top-k windows

typedef __attribute__((ext_vector_type(16))) __bf16 v16bf;
typedef __attribute__((ext_vector_type(8)))  float  v8f;
typedef __attribute__((ext_vector_type(4)))  int    v4i;

#define AS1 __attribute__((address_space(1)))
#define AS3 __attribute__((address_space(3)))

#if __has_builtin(__builtin_amdgcn_global_load_async_to_lds_b128) && \
    __has_builtin(__builtin_amdgcn_s_wait_asynccnt)
#define USE_ASYNC_LDS 1
#else
#define USE_ASYNC_LDS 0
#endif

// copy 16 bytes global -> LDS (async on gfx1250 when the builtin exists)
__device__ __forceinline__ void copy16_g2l(const __bf16* g, __bf16* l) {
#if USE_ASYNC_LDS
  // signature (from hipcc diagnostic): (int4 AS1*, int4 AS3*, imm offset, imm cpol)
  __builtin_amdgcn_global_load_async_to_lds_b128((AS1 v4i*)g, (AS3 v4i*)l, 0, 0);
#else
  *(uint4*)l = *(const uint4*)g;
#endif
}

// ---------------- WMMA helpers (wave32, V_WMMA_F32_16X16X32_BF16) -----------
// A fragment (16x32, MxK): lane l -> M = l&15, half h = l>>4.
// element e: K = e + 16*h
__device__ __forceinline__ int frag_a_k(int e, int h) {
  return e + (h << 4);
}

template <typename T>
__device__ __forceinline__ v16bf load_a_rowmajor(const T* tile, int ld, int lane) {
  const int m = lane & 15, h = lane >> 4;
  v16bf a;
#pragma unroll
  for (int e = 0; e < 16; ++e) {
    const int k = frag_a_k(e, h);
    a[e] = (__bf16)tile[m * ld + k];
  }
  return a;
}

// B fragment (32x16, KxN): lane l -> N = l&15, half h = l>>4; element e -> K = e + 16*h.
// Source layout: column n contiguous over k with stride ld  (B[k][n] = colbase[n*ld + k])
template <typename T>
__device__ __forceinline__ v16bf load_b_colmajor(const T* colbase, int ld, int lane) {
  const int n = lane & 15, h = lane >> 4;
  v16bf b;
#pragma unroll
  for (int e = 0; e < 16; ++e) {
    const int k = e + (h << 4);
    b[e] = (__bf16)colbase[n * ld + k];
  }
  return b;
}

// B fragment from an LDS-staged 16x32 bf16 tile (row n contiguous, 32 per row)
__device__ __forceinline__ v16bf load_b_lds(const __bf16* buf, int lane) {
  const int n = lane & 15, h = lane >> 4;
  v16bf b;
#pragma unroll
  for (int e = 0; e < 16; ++e) b[e] = buf[n * 32 + (h << 4) + e];
  return b;
}

__device__ __forceinline__ v8f wmma_bf16(v16bf a, v16bf b, v8f c) {
  // 8 args: (neg_a, A, neg_b, B, c_mod, C, reuse_a, reuse_b)
  return __builtin_amdgcn_wmma_f32_16x16x32_bf16(false, a, false, b, (short)0, c,
                                                 false, false);
}

// ---------------- K0: f32 -> bf16 converter (x4 per thread) -----------------
__global__ void cvt_bf16_kernel(const float* __restrict__ x, __bf16* __restrict__ y,
                                int n) {
  const int i = (blockIdx.x * blockDim.x + threadIdx.x) * 4;
  if (i + 3 < n) {
    const float4 v = *(const float4*)(x + i);
    y[i + 0] = (__bf16)v.x;
    y[i + 1] = (__bf16)v.y;
    y[i + 2] = (__bf16)v.z;
    y[i + 3] = (__bf16)v.w;
  }
}

// ---------------- K1 / K6: Y = X @ W^T + bias --------------------------------
// Block = 8 waves covering 8 M-tiles of ONE N-tile; the 16x32 bf16 weight slab
// is staged once per block into LDS (async, double-buffered) and shared by all
// 8 waves -> 8x less weight traffic than per-wave loads.
__global__ void __launch_bounds__(256)
gemm_bf16_kernel(const __bf16* __restrict__ A,    // [M,K] bf16
                 const __bf16* __restrict__ Wb,   // [N,K] bf16 (row-major)
                 const float* __restrict__ bias,  // [N]
                 float* __restrict__ Y,           // [M,N] f32
                 int M, int N, int K) {
  __shared__ __bf16 wbuf[2][16 * 32];  // double-buffered B slab (1 KB each)

  const int tid  = threadIdx.x;
  const int lane = tid & 31;
  const int wv   = tid >> 5;  // wave 0..7 -> M-tile within group
  const int ntiles = N >> 4;
  const int nt = blockIdx.x % ntiles;
  const int mg = blockIdx.x / ntiles;
  const int mt = mg * 8 + wv;
  if (mt * 16 >= M) return;

  // stage one 16x32 bf16 weight slab: 16 rows x 64B = 64 16B-chunks (waves 0-1)
  auto stage = [&](int buf, int k0) {
    if (tid < 64) {
      const int r = tid >> 2;        // weight row 0..15
      const int c = (tid & 3) * 8;   // bf16 col 0,8,16,24
      copy16_g2l(Wb + (size_t)(nt * 16 + r) * K + k0 + c, &wbuf[buf][r * 32 + c]);
    }
  };

  stage(0, 0);

  const size_t arow = (size_t)mt * 16 * K;
  v8f acc = {};
  for (int k0 = 0, it = 0; k0 < K; k0 += 32, ++it) {
    const int cur = it & 1;
    if (k0 + 32 < K) {
      stage(cur ^ 1, k0 + 32);     // prefetch next slab into other buffer
#if USE_ASYNC_LDS
      __builtin_amdgcn_s_wait_asynccnt(1);  // current slab resident
#endif
    } else {
#if USE_ASYNC_LDS
      __builtin_amdgcn_s_wait_asynccnt(0);  // drain
#endif
    }
    __syncthreads();               // slab visible to all 8 waves

    v16bf a = load_a_rowmajor(A + arow + k0, K, lane);
    v16bf b = load_b_lds(wbuf[cur], lane);
    acc = wmma_bf16(a, b, acc);

    __syncthreads();               // everyone done reading before restage
  }

  const int n = lane & 15, h = lane >> 4;
  const float bv = bias[nt * 16 + n];
#pragma unroll
  for (int j = 0; j < 8; ++j) {
    const int m = j + (h << 3);
    Y[(size_t)(mt * 16 + m) * N + nt * 16 + n] = acc[j] + bv;
  }
}

// ---------------- K2: split heads + L2 normalize (wave per (b,h,s)) ---------
__global__ void headsplit_norm_kernel(const float* __restrict__ qf,
                                      const float* __restrict__ kf,
                                      const float* __restrict__ vf,
                                      __bf16* __restrict__ qhb,  // [B,H,S,DH]
                                      __bf16* __restrict__ qnb,
                                      __bf16* __restrict__ khb,
                                      __bf16* __restrict__ kub,
                                      __bf16* __restrict__ vhb) {
  const int r = blockIdx.x;  // r = (b*H + h)*S + s
  const int b = r / (HD * SEQ);
  const int h = (r / SEQ) % HD;
  const int s = r % SEQ;
  const int lane = threadIdx.x;  // 32 lanes, 2 elems each
  const size_t src = (size_t)(b * SEQ + s) * EMB + h * DH;
  const size_t dst = (size_t)r * DH;

  const float q0 = qf[src + lane], q1 = qf[src + lane + 32];
  const float k0 = kf[src + lane], k1 = kf[src + lane + 32];
  const float v0 = vf[src + lane], v1 = vf[src + lane + 32];

  float qs = q0 * q0 + q1 * q1;
  float ks = k0 * k0 + k1 * k1;
#pragma unroll
  for (int m = 16; m; m >>= 1) {
    qs += __shfl_xor(qs, m, 32);
    ks += __shfl_xor(ks, m, 32);
  }
  const float qinv = 1.f / fmaxf(sqrtf(qs), 1e-12f);
  const float kinv = 1.f / fmaxf(sqrtf(ks), 1e-12f);

  qhb[dst + lane] = (__bf16)q0;          qhb[dst + lane + 32] = (__bf16)q1;
  qnb[dst + lane] = (__bf16)(q0 * qinv); qnb[dst + lane + 32] = (__bf16)(q1 * qinv);
  khb[dst + lane] = (__bf16)k0;          khb[dst + lane + 32] = (__bf16)k1;
  kub[dst + lane] = (__bf16)(k0 * kinv); kub[dst + lane + 32] = (__bf16)(k1 * kinv);
  vhb[dst + lane] = (__bf16)v0;          vhb[dst + lane + 32] = (__bf16)v1;
}

// ---------------- K3: coarse scores: sum_s (qn.k_unit)^2 / 64 / t -----------
// one wave per (b, h, q-tile, window)
__global__ void coarse_kernel(const __bf16* __restrict__ qnb,
                              const __bf16* __restrict__ kub,
                              const float* __restrict__ temp,
                              float* __restrict__ scores) {  // [B,H,S,NW]
  const int lane = threadIdx.x & 31;
  const int wid  = (blockIdx.x * blockDim.x + threadIdx.x) >> 5;
  const int w  = wid & (NW - 1);
  const int qt = (wid >> 3) & 31;
  const int h  = (wid >> 8) & (HD - 1);
  const int b  = wid >> 11;

  const size_t headbase = (size_t)(b * HD + h) * SEQ;
  const __bf16* qtile = qnb + (headbase + qt * 16) * DH;
  const v16bf a0 = load_a_rowmajor(qtile, DH, lane);
  const v16bf a1 = load_a_rowmajor(qtile + 32, DH, lane);

  float sums[8] = {0, 0, 0, 0, 0, 0, 0, 0};
#pragma unroll
  for (int st = 0; st < 4; ++st) {  // 4 x 16 keys = one window
    const __bf16* colb = kub + (headbase + w * WSZ + st * 16) * DH;
    v8f acc = {};
    acc = wmma_bf16(a0, load_b_colmajor(colb, DH, lane), acc);
    acc = wmma_bf16(a1, load_b_colmajor(colb + 32, DH, lane), acc);
#pragma unroll
    for (int j = 0; j < 8; ++j) sums[j] += acc[j] * acc[j];
  }
  // reduce across the 16 lanes holding the 16 key-columns of each row group
#pragma unroll
  for (int j = 0; j < 8; ++j) {
#pragma unroll
    for (int m = 1; m <= 8; m <<= 1) sums[j] += __shfl_xor(sums[j], m, 32);
  }
  const float t = fmaxf(temp[0], 0.1f);
  const float f = 1.f / (64.f * t);  // wts = 1/64 (S divisible by WSZ -> no pad)
  if ((lane & 15) == 0) {
    const int hh = lane >> 4;
#pragma unroll
    for (int j = 0; j < 8; ++j) {
      const int q = qt * 16 + j + (hh << 3);
      scores[(headbase + q) * NW + w] = sums[j] * f;
    }
  }
}

// ---------------- K4: top-4 of 8 + log-softmax of coarse weights ------------
__global__ void topk_kernel(const float* __restrict__ scores,
                            int* __restrict__ topidx, float* __restrict__ logc) {
  const int q = blockIdx.x * blockDim.x + threadIdx.x;  // over B*H*S
  if (q >= BATCH * HD * SEQ) return;
  float sc[NW];
#pragma unroll
  for (int w = 0; w < NW; ++w) sc[w] = scores[(size_t)q * NW + w];
  int   idx[NTOP];
  float val[NTOP];
#pragma unroll
  for (int t = 0; t < NTOP; ++t) {
    int bi = 0;
    float bv = sc[0];
#pragma unroll
    for (int w = 1; w < NW; ++w)
      if (sc[w] > bv) { bv = sc[w]; bi = w; }
    idx[t] = bi; val[t] = bv; sc[bi] = -1e30f;
  }
  const float mx = val[0];
  float sum = 0.f;
#pragma unroll
  for (int t = 0; t < NTOP; ++t) sum += __expf(val[t] - mx);
  const float ls = __logf(sum);
#pragma unroll
  for (int t = 0; t < NTOP; ++t) {
    topidx[(size_t)q * NTOP + t] = idx[t];
    logc[(size_t)q * NTOP + t]   = fmaxf(val[t] - mx - ls, -18.420681f);  // log(1e-8)
  }
}

// ---------------- K5: fine attention -----------------------------------------
// Exact reformulation of the per-query top-k gather: attention over ALL 512
// keys with additive bias  bias[q][w] = log_coarse (selected) / -inf (masked);
// masked windows get probability 0, identical to the gathered reference, and
// the 16-query WMMA tile keeps a shared key set. One wave per (b,h,q-tile).
__global__ void __launch_bounds__(32)
fine_attn_kernel(const __bf16* __restrict__ qhb, const __bf16* __restrict__ khb,
                 const __bf16* __restrict__ vhb, const int* __restrict__ topidx,
                 const float* __restrict__ logc, __bf16* __restrict__ attnout) {
  __shared__ float sc[16 * SEQ];
  __shared__ float biasw[16][NW];
  __shared__ float rinv[16];

  const int lane = threadIdx.x;
  const int qt = blockIdx.x & 31;
  const int h  = (blockIdx.x >> 5) & (HD - 1);
  const int b  = blockIdx.x >> 8;
  const size_t headbase = (size_t)(b * HD + h) * SEQ;

  if (lane < 16) {
    const int q = qt * 16 + lane;
    float bb[NW];
#pragma unroll
    for (int w = 0; w < NW; ++w) bb[w] = -1e30f;
    const size_t base = (headbase + q) * NTOP;
#pragma unroll
    for (int t = 0; t < NTOP; ++t) bb[topidx[base + t]] = logc[base + t];
#pragma unroll
    for (int w = 0; w < NW; ++w) biasw[lane][w] = bb[w];
  }
  __syncthreads();

  const __bf16* qtile = qhb + (headbase + qt * 16) * DH;
  const v16bf a0 = load_a_rowmajor(qtile, DH, lane);
  const v16bf a1 = load_a_rowmajor(qtile + 32, DH, lane);
  const float scale = 0.125f;  // d^-0.5, d = 64
  const int n = lane & 15, hh = lane >> 4;

  // scores = q @ k^T * scale + window bias
  for (int st = 0; st < SEQ / 16; ++st) {
    const __bf16* colb = khb + (headbase + st * 16) * DH;
    v8f acc = {};
    acc = wmma_bf16(a0, load_b_colmajor(colb, DH, lane), acc);
    acc = wmma_bf16(a1, load_b_colmajor(colb + 32, DH, lane), acc);
    const int w = st >> 2;  // 4 key-tiles per window
#pragma unroll
    for (int j = 0; j < 8; ++j) {
      const int m = j + (hh << 3);
      sc[m * SEQ + st * 16 + n] = acc[j] * scale + biasw[m][w];
    }
  }
  __syncthreads();

  // row softmax; store p into sc, keep 1/sum
  for (int r = 0; r < 16; ++r) {
    float mx = -1e30f;
    for (int i = lane; i < SEQ; i += 32) mx = fmaxf(mx, sc[r * SEQ + i]);
#pragma unroll
    for (int m2 = 16; m2; m2 >>= 1) mx = fmaxf(mx, __shfl_xor(mx, m2, 32));
    float sum = 0.f;
    for (int i = lane; i < SEQ; i += 32) {
      const float p = __expf(sc[r * SEQ + i] - mx);
      sc[r * SEQ + i] = p;
      sum += p;
    }
#pragma unroll
    for (int m2 = 16; m2; m2 >>= 1) sum += __shfl_xor(sum, m2, 32);
    if (lane == 0) rinv[r] = 1.f / sum;
  }
  __syncthreads();

  // out = P @ V   (K = 512 keys, N = 64 head dims)
  v8f accv[4] = {};
  for (int ks = 0; ks < SEQ / 32; ++ks) {
    v16bf pa;
    const int m = lane & 15;
    const float ri = rinv[m];
#pragma unroll
    for (int e = 0; e < 16; ++e) {
      const int k = frag_a_k(e, hh);
      pa[e] = (__bf16)(sc[m * SEQ + ks * 32 + k] * ri);
    }
    const __bf16* vrow = vhb + (headbase + ks * 32) * DH;
#pragma unroll
    for (int nt = 0; nt < 4; ++nt) {
      v16bf vb;
#pragma unroll
      for (int e = 0; e < 16; ++e) {
        const int k = e + (hh << 4);  // key within this 32-slab
        vb[e] = vrow[(size_t)k * DH + nt * 16 + n];
      }
      accv[nt] = wmma_bf16(pa, vb, accv[nt]);
    }
  }

  // write context in [B*S, EMB] row-major (bf16) for the output GEMM
#pragma unroll
  for (int nt = 0; nt < 4; ++nt) {
#pragma unroll
    for (int j = 0; j < 8; ++j) {
      const int m = j + (hh << 3);
      const size_t row = (size_t)b * SEQ + qt * 16 + m;
      const int col = h * DH + nt * 16 + n;
      attnout[row * EMB + col] = (__bf16)accv[nt][j];
    }
  }
}

// ---------------- host-side orchestration -----------------------------------
extern "C" void kernel_launch(void* const* d_in, const int* in_sizes, int n_in,
                              void* d_out, int out_size, void* d_ws, size_t ws_size,
                              hipStream_t stream) {
  (void)in_sizes; (void)n_in; (void)out_size; (void)ws_size;
  const float* query = (const float*)d_in[0];
  const float* key   = (const float*)d_in[1];
  const float* value = (const float*)d_in[2];
  const float* Wq = (const float*)d_in[3];  const float* bq = (const float*)d_in[4];
  const float* Wk = (const float*)d_in[5];  const float* bk = (const float*)d_in[6];
  const float* Wv = (const float*)d_in[7];  const float* bv = (const float*)d_in[8];
  const float* Wo = (const float*)d_in[9];  const float* bo = (const float*)d_in[10];
  const float* temp = (const float*)d_in[11];

  const int M = BATCH * SEQ;  // 1024
  const int N = EMB;          // 512
  const int K = EMB;          // 512

  char* ws = (char*)d_ws;
  size_t off = 0;
  const size_t f32_bse = (size_t)M * EMB * sizeof(float);                 // 2 MB
  const size_t bf_head = (size_t)BATCH * HD * SEQ * DH * sizeof(__bf16);  // 1 MB
  const size_t bf_x    = (size_t)M * EMB * sizeof(__bf16);                // 1 MB
  const size_t bf_w    = (size_t)EMB * EMB * sizeof(__bf16);              // 512 KB

  float*  qf  = (float*)(ws + off);  off += f32_bse;
  float*  kf  = (float*)(ws + off);  off += f32_bse;
  float*  vf  = (float*)(ws + off);  off += f32_bse;
  __bf16* qhb = (__bf16*)(ws + off); off += bf_head;
  __bf16* qnb = (__bf16*)(ws + off); off += bf_head;
  __bf16* khb = (__bf16*)(ws + off); off += bf_head;
  __bf16* kub = (__bf16*)(ws + off); off += bf_head;
  __bf16* vhb = (__bf16*)(ws + off); off += bf_head;
  float*  scores = (float*)(ws + off);  off += (size_t)BATCH * HD * SEQ * NW * sizeof(float);
  int*    tidx   = (int*)(ws + off);    off += (size_t)BATCH * HD * SEQ * NTOP * sizeof(int);
  float*  logc   = (float*)(ws + off);  off += (size_t)BATCH * HD * SEQ * NTOP * sizeof(float);
  __bf16* attnbf = (__bf16*)(ws + off); off += bf_x;
  __bf16* xqb = (__bf16*)(ws + off); off += bf_x;
  __bf16* xkb = (__bf16*)(ws + off); off += bf_x;
  __bf16* xvb = (__bf16*)(ws + off); off += bf_x;
  __bf16* wqb = (__bf16*)(ws + off); off += bf_w;
  __bf16* wkb = (__bf16*)(ws + off); off += bf_w;
  __bf16* wvb = (__bf16*)(ws + off); off += bf_w;
  __bf16* wob = (__bf16*)(ws + off); off += bf_w;

  // K0: one-time f32 -> bf16 conversions (activations + weights)
  const int nx = M * EMB, nw = EMB * EMB;
  cvt_bf16_kernel<<<nx / 1024, 256, 0, stream>>>(query, xqb, nx);
  cvt_bf16_kernel<<<nx / 1024, 256, 0, stream>>>(key,   xkb, nx);
  cvt_bf16_kernel<<<nx / 1024, 256, 0, stream>>>(value, xvb, nx);
  cvt_bf16_kernel<<<nw / 1024, 256, 0, stream>>>(Wq, wqb, nw);
  cvt_bf16_kernel<<<nw / 1024, 256, 0, stream>>>(Wk, wkb, nw);
  cvt_bf16_kernel<<<nw / 1024, 256, 0, stream>>>(Wv, wvb, nw);
  cvt_bf16_kernel<<<nw / 1024, 256, 0, stream>>>(Wo, wob, nw);

  // K1: q/k/v projections — block = 8 M-tiles x 1 N-tile sharing LDS weight slab
  const int gemm_blocks = (M / 16 / 8) * (N / 16);  // 256
  gemm_bf16_kernel<<<gemm_blocks, 256, 0, stream>>>(xqb, wqb, bq, qf, M, N, K);
  gemm_bf16_kernel<<<gemm_blocks, 256, 0, stream>>>(xkb, wkb, bk, kf, M, N, K);
  gemm_bf16_kernel<<<gemm_blocks, 256, 0, stream>>>(xvb, wvb, bv, vf, M, N, K);

  // K2: head split + normalization
  headsplit_norm_kernel<<<BATCH * HD * SEQ, 32, 0, stream>>>(qf, kf, vf, qhb, qnb,
                                                             khb, kub, vhb);

  // K3: coarse window scores — 4096 waves
  coarse_kernel<<<512, 256, 0, stream>>>(qnb, kub, temp, scores);

  // K4: top-k + log-softmax
  topk_kernel<<<(BATCH * HD * SEQ + 255) / 256, 256, 0, stream>>>(scores, tidx, logc);

  // K5: fine (routed) attention — one wave per (b,h,q-tile)
  fine_attn_kernel<<<BATCH * HD * (SEQ / 16), 32, 0, stream>>>(qhb, khb, vhb, tidx,
                                                               logc, attnbf);

  // K6: output projection -> d_out (f32)
  gemm_bf16_kernel<<<gemm_blocks, 256, 0, stream>>>(attnbf, wob, bo, (float*)d_out,
                                                    M, N, K);
}
